// Pointnet2_84293028152073
// MI455X (gfx1250) — matrix operand order repaired and checked by reading, not compiled, not run on hardware
//
#include <hip/hip_runtime.h>
#include <hip/hip_bf16.h>

typedef __attribute__((ext_vector_type(16))) __bf16 v16bf;
typedef __attribute__((ext_vector_type(8)))  float  v8f;
typedef __attribute__((ext_vector_type(2)))  float  f32x2;
typedef __attribute__((ext_vector_type(2)))  __bf16 bf16x2;

union FragB { unsigned int u[8]; v16bf v; };

// hardware packed convert: one v_cvt_pk_bf16_f32 per pair
__device__ __forceinline__ unsigned int pack_bf16x2(float lo, float hi) {
    union { bf16x2 v; unsigned int u; } p;
    f32x2 t; t.x = lo; t.y = hi;
    p.v = __builtin_convertvector(t, bf16x2);
    return p.u;
}
__device__ __forceinline__ unsigned short f32_to_bf16_bits(float f) {
    union { bf16x2 v; unsigned short s[2]; } p;
    f32x2 t; t.x = f; t.y = f;
    p.v = __builtin_convertvector(t, bf16x2);
    return p.s[0];
}
__device__ __forceinline__ float bf16_to_f32(unsigned short h) {
    union { unsigned int u; float f; } c; c.u = ((unsigned int)h) << 16;
    return c.f;
}

// ---------------------------------------------------------------------------
// GEMM  Y[D, B*N] = W[D, C] * X[C, B*N]  (+bias), bf16 inputs, f32 accumulate.
// Fuses previous layer's BN+ReLU into the X load (FIRST=false) and accumulates
// per-channel sum / sumsq for the next BN.
// Tile: 64(M) x 128(cols) x 32(K), 8 waves (4xM, 2xN), 4 WMMA accums/wave.
// Double-buffered LDS + register staging: one barrier per K-step, next tile's
// global loads overlap the WMMAs.
// ---------------------------------------------------------------------------
template <bool FIRST>
__global__ __launch_bounds__(256)
void gemm_bn(const float* __restrict__ W, const float* __restrict__ bias,
             const float* __restrict__ Xf, const unsigned short* __restrict__ Xb,
             const float* __restrict__ sIn, const float* __restrict__ tIn,
             unsigned short* __restrict__ Y,
             float* __restrict__ sumOut, float* __restrict__ sqOut,
             int C, int D)
{
    const int N = 4096;
    __shared__ unsigned short As[2][64 * 32];     // [m][k]   2 x 4 KB
    __shared__ unsigned short Bs[2][128 * 32];    // [col][k] 2 x 8 KB (transposed)

    const int tid  = threadIdx.x;
    const int lane = tid & 31;
    const int wave = tid >> 5;
    const int wm   = wave & 3;
    const int wn   = wave >> 2;

    const int       m0      = blockIdx.y * 64;
    const long long colBase = (long long)blockIdx.x * 128;
    const int       b       = (int)(colBase >> 12);    // p / 4096
    const int       n0      = (int)(colBase & 4095);

    v8f acc[4] = {};

    // cooperative fill index maps
    const int am = tid >> 2;            // 0..63  (A row)
    const int ak = (tid & 3) * 8;       // 0,8,16,24
    const int bj = tid & 127;           // 0..127 (B col)
    const int bk = (tid >> 7) * 16;     // 0 or 16

    const float*     wp    = W + (long long)(m0 + am) * C + ak;  // wp[k0+i]
    const long long  xbase = (long long)b * C * N + n0 + bj;     // + kg*N

    const int KF = C >> 5;              // full 32-wide K steps (no guards)

    float aw[8];
    float xv[16];

    // ---- register-stage fetch of one full K tile (unguarded) ----
    auto fetch = [&](int k0) {
        #pragma unroll
        for (int i = 0; i < 8; ++i) aw[i] = wp[k0 + i];
        #pragma unroll
        for (int i = 0; i < 16; ++i) {
            const int kg = k0 + bk + i;
            if (FIRST) {
                xv[i] = Xf[xbase + (long long)kg * N];
            } else {
                const float v = bf16_to_f32(Xb[xbase + (long long)kg * N]);
                xv[i] = fmaxf(fmaf(sIn[kg], v, tIn[kg]), 0.0f);
            }
        }
    };
    // ---- pack staged registers into LDS as bf16 (v_cvt_pk + b128 stores) ----
    auto stage = [&](int buf) {
        uint4 wv;
        wv.x = pack_bf16x2(aw[0], aw[1]);  wv.y = pack_bf16x2(aw[2], aw[3]);
        wv.z = pack_bf16x2(aw[4], aw[5]);  wv.w = pack_bf16x2(aw[6], aw[7]);
        *(uint4*)&As[buf][am * 32 + ak] = wv;
        uint4 x0, x1;
        x0.x = pack_bf16x2(xv[0],  xv[1]);  x0.y = pack_bf16x2(xv[2],  xv[3]);
        x0.z = pack_bf16x2(xv[4],  xv[5]);  x0.w = pack_bf16x2(xv[6],  xv[7]);
        x1.x = pack_bf16x2(xv[8],  xv[9]);  x1.y = pack_bf16x2(xv[10], xv[11]);
        x1.z = pack_bf16x2(xv[12], xv[13]); x1.w = pack_bf16x2(xv[14], xv[15]);
        *(uint4*)&Bs[buf][bj * 32 + bk]     = x0;
        *(uint4*)&Bs[buf][bj * 32 + bk + 8] = x1;
    };
    // ---- fragment loads per ISA 7.12.2 + 4x WMMA ----
    auto compute = [&](int buf) {
        FragB a;
        const int arow = wm * 16 + (lane & 15);
        const int akb  = (lane >> 4) * 8;
        #pragma unroll
        for (int h = 0; h < 8; ++h) {
            const int k = ((h & 4) ? 16 : 0) + akb + 2 * (h & 3);
            a.u[h] = *(const unsigned int*)&As[buf][arow * 32 + k];
        }
        const int bkb = (lane >> 4) * 8;
        #pragma unroll
        for (int f = 0; f < 4; ++f) {
            FragB bb;
            const int j = wn * 64 + f * 16 + (lane & 15);
            #pragma unroll
            for (int h = 0; h < 8; ++h)
                bb.u[h] = *(const unsigned int*)&Bs[buf][j * 32 + ((h & 4) ? 16 : 0) + bkb + 2 * (h & 3)];
            acc[f] = __builtin_amdgcn_wmma_f32_16x16x32_bf16(
                false, a.v, false, bb.v, (short)0, acc[f], false, false);
        }
    };

    // ---- software-pipelined main loop: 1 barrier / K-step ----
    fetch(0);
    stage(0);
    for (int kt = 0; kt < KF; ++kt) {
        __syncthreads();
        if (kt + 1 < KF) fetch((kt + 1) << 5);    // global loads overlap WMMAs
        compute(kt & 1);
        if (kt + 1 < KF) stage((kt + 1) & 1);
    }

    // ---- guarded tail step (layer 1 only: 643 = 20*32 + 3) ----
    if (C & 31) {
        const int k0 = KF << 5;
        __syncthreads();
        #pragma unroll
        for (int i = 0; i < 8; ++i) {
            const int kg = k0 + ak + i;
            aw[i] = (kg < C) ? wp[k0 + i] : 0.0f;
        }
        #pragma unroll
        for (int i = 0; i < 16; ++i) {
            const int kg = k0 + bk + i;
            float x = 0.0f;
            if (kg < C) {
                if (FIRST) x = Xf[xbase + (long long)kg * N];
                else {
                    const float v = bf16_to_f32(Xb[xbase + (long long)kg * N]);
                    x = fmaxf(fmaf(sIn[kg], v, tIn[kg]), 0.0f);
                }
            }
            xv[i] = x;
        }
        stage(KF & 1);
        __syncthreads();
        compute(KF & 1);
    }

    // ---- epilogue: +bias, bf16 store, per-channel sum / sumsq atomics ----
    const int nloc = lane & 15;
    const int hs   = lane >> 4;      // C/D layout: lane<16 -> M=r ; lane>=16 -> M=8+r
    #pragma unroll
    for (int r = 0; r < 8; ++r) {
        const int   mg = m0 + wm * 16 + hs * 8 + r;
        const float bv = bias[mg];
        float p1 = 0.0f, p2 = 0.0f;
        #pragma unroll
        for (int f = 0; f < 4; ++f) {
            const float y = acc[f][r] + bv;
            const long long p = colBase + wn * 64 + f * 16 + nloc;
            const int n = (int)(p & 4095);
            Y[((long long)b * D + mg) * (long long)N + n] = f32_to_bf16_bits(y);
            p1 += y;  p2 += y * y;
        }
        #pragma unroll
        for (int m = 1; m < 16; m <<= 1) {   // reduce over 16 lanes sharing row mg
            p1 += __shfl_xor(p1, m);
            p2 += __shfl_xor(p2, m);
        }
        if (nloc == 0) {
            atomicAdd(&sumOut[mg], p1);
            atomicAdd(&sqOut[mg],  p2);
        }
    }
}

// sum/sumsq -> per-channel affine: s = g*rsqrt(var+eps), t = be - mean*s
__global__ void bn_stats(const float* __restrict__ sum, const float* __restrict__ sq,
                         const float* __restrict__ g, const float* __restrict__ be,
                         float* __restrict__ sOut, float* __restrict__ tOut,
                         int D, float invCnt)
{
    const int c = blockIdx.x * blockDim.x + threadIdx.x;
    if (c >= D) return;
    const float mean = sum[c] * invCnt;
    const float var  = sq[c] * invCnt - mean * mean;
    const float s    = g[c] * rsqrtf(var + 1e-5f);
    sOut[c] = s;
    tOut[c] = be[c] - mean * s;
}

// one wave per (b,ch): pooled = max_n relu(s*y + t); track max & min so the
// monotone-map trick is valid for either sign of s.
__global__ __launch_bounds__(256)
void pool_max(const unsigned short* __restrict__ Y3,
              const float* __restrict__ s3, const float* __restrict__ t3,
              float* __restrict__ pooled)
{
    const int D = 1024, N = 4096;
    const int gw   = (blockIdx.x * 256 + threadIdx.x) >> 5;
    const int lane = threadIdx.x & 31;
    if (gw >= 16 * D) return;
    const int b = gw / D, ch = gw % D;
    const long long base = ((long long)b * D + ch) * N;
    float mx = -3.402823466e+38f, mn = 3.402823466e+38f;
    for (int i = lane; i < N; i += 32) {
        const float v = bf16_to_f32(Y3[base + i]);
        mx = fmaxf(mx, v);  mn = fminf(mn, v);
    }
    #pragma unroll
    for (int m = 1; m < 32; m <<= 1) {
        mx = fmaxf(mx, __shfl_xor(mx, m));
        mn = fminf(mn, __shfl_xor(mn, m));
    }
    if (lane == 0) {
        const float s = s3[ch], t = t3[ch];
        const float h = (s >= 0.0f) ? mx : mn;
        pooled[(long long)b * D + ch] = fmaxf(fmaf(s, h, t), 0.0f);
    }
}

// final BatchNorm1d over the batch dimension (16 samples per channel)
__global__ void final_bn(const float* __restrict__ pooled,
                         const float* __restrict__ g4, const float* __restrict__ be4,
                         float* __restrict__ out)
{
    const int B = 16, D = 1024;
    const int ch = blockIdx.x * blockDim.x + threadIdx.x;
    if (ch >= D) return;
    float v[16]; float sum = 0.0f;
    #pragma unroll
    for (int b = 0; b < B; ++b) { v[b] = pooled[b * D + ch]; sum += v[b]; }
    const float mean = sum * (1.0f / 16.0f);
    float var = 0.0f;
    #pragma unroll
    for (int b = 0; b < B; ++b) { const float d = v[b] - mean; var += d * d; }
    var *= (1.0f / 16.0f);
    const float inv = rsqrtf(var + 1e-5f);
    const float g = g4[ch], be = be4[ch];
    #pragma unroll
    for (int b = 0; b < B; ++b)
        out[b * D + ch] = fmaf(g * (v[b] - mean), inv, be);
}

extern "C" void kernel_launch(void* const* d_in, const int* in_sizes, int n_in,
                              void* d_out, int out_size, void* d_ws, size_t ws_size,
                              hipStream_t stream) {
    const float* xyz = (const float*)d_in[0];
    const float* W1  = (const float*)d_in[1];
    const float* b1  = (const float*)d_in[2];
    const float* g1  = (const float*)d_in[3];
    const float* be1 = (const float*)d_in[4];
    const float* W2  = (const float*)d_in[5];
    const float* b2  = (const float*)d_in[6];
    const float* g2  = (const float*)d_in[7];
    const float* be2 = (const float*)d_in[8];
    const float* W3  = (const float*)d_in[9];
    const float* b3  = (const float*)d_in[10];
    const float* g3  = (const float*)d_in[11];
    const float* be3 = (const float*)d_in[12];
    const float* g4  = (const float*)d_in[13];
    const float* be4 = (const float*)d_in[14];
    float* out = (float*)d_out;

    const int Bsz = 16, N = 4096;
    const float invCnt = 1.0f / (float)(Bsz * N);   // 1/65536

    // workspace carve-up
    char* ws = (char*)d_ws;
    unsigned short* Y1 = (unsigned short*)ws;  ws += (size_t)Bsz * 256  * N * 2;  // 32 MB
    unsigned short* Y2 = (unsigned short*)ws;  ws += (size_t)Bsz * 512  * N * 2;  // 64 MB
    unsigned short* Y3 = (unsigned short*)ws;  ws += (size_t)Bsz * 1024 * N * 2;  // 128 MB
    float* statBase = (float*)ws;
    float* sum1 = statBase;        float* sq1 = sum1 + 256;
    float* sum2 = sq1  + 256;      float* sq2 = sum2 + 512;
    float* sum3 = sq2  + 512;      float* sq3 = sum3 + 1024;
    ws += (256 + 256 + 512 + 512 + 1024 + 1024) * sizeof(float);
    float* s1 = (float*)ws;  float* t1 = s1 + 256;
    float* s2 = t1 + 256;    float* t2 = s2 + 512;
    float* s3 = t2 + 512;    float* t3 = s3 + 1024;
    ws += (256 + 256 + 512 + 512 + 1024 + 1024) * sizeof(float);
    float* pooled = (float*)ws;  // 16*1024 f32

    // zero stat accumulators every call (graph-capture-safe async memset)
    hipMemsetAsync(statBase, 0, (256 + 256 + 512 + 512 + 1024 + 1024) * sizeof(float), stream);

    const dim3 blk(256);
    // layer 1: 643 -> 256 (raw f32 input, no fused BN)
    gemm_bn<true ><<<dim3(512,  4), blk, 0, stream>>>(W1, b1, xyz, nullptr, nullptr, nullptr,
                                                      Y1, sum1, sq1, 643, 256);
    bn_stats<<<dim3(1),  blk, 0, stream>>>(sum1, sq1, g1, be1, s1, t1, 256,  invCnt);
    // layer 2: 256 -> 512 (fused BN1+ReLU on load)
    gemm_bn<false><<<dim3(512,  8), blk, 0, stream>>>(W2, b2, nullptr, Y1, s1, t1,
                                                      Y2, sum2, sq2, 256, 512);
    bn_stats<<<dim3(2),  blk, 0, stream>>>(sum2, sq2, g2, be2, s2, t2, 512,  invCnt);
    // layer 3: 512 -> 1024 (fused BN2+ReLU on load)
    gemm_bn<false><<<dim3(512, 16), blk, 0, stream>>>(W3, b3, nullptr, Y2, s2, t2,
                                                      Y3, sum3, sq3, 512, 1024);
    bn_stats<<<dim3(4),  blk, 0, stream>>>(sum3, sq3, g3, be3, s3, t3, 1024, invCnt);
    // BN3+ReLU fused into max-pool over N
    pool_max<<<dim3(2048), blk, 0, stream>>>(Y3, s3, t3, pooled);
    // final batch-dim BN -> d_out [16,1024]
    final_bn<<<dim3(4),  blk, 0, stream>>>(pooled, g4, be4, out);
}
